// MultiHeadAttention_6322191859742
// MI455X (gfx1250) — matrix operand order repaired and checked by reading, not compiled, not run on hardware
//
#include <hip/hip_runtime.h>

// ---------------------------------------------------------------------------
// MoE-LoRA multi-head attention for MI455X (gfx1250, wave32, WMMA f16).
// Heavy GEMMs: v_wmma_f32_16x16x32_f16, 64x64 block tile, double-buffered
// LDS with ASYNCcnt-tracked global_load_async_to_lds_b128 when available.
// qk softmax is online (1 stat pass + 1 weight pass over the 256MB tensor),
// fused with w@V WMMA.
// ---------------------------------------------------------------------------

typedef _Float16 v8h  __attribute__((ext_vector_type(8)));
typedef _Float16 v16h __attribute__((ext_vector_type(16)));
typedef float    v8f  __attribute__((ext_vector_type(8)));
typedef int      v4i  __attribute__((ext_vector_type(4)));

#define BATCH    4
#define SEQ      1024
#define D_MODEL  1024
#define NHEAD    16
#define DHEAD    64
#define NEXP     8
#define RANK     8
#define LORA_SCALE 2.0f
#define QK_SCALE   0.35355339059327373f   // 64^-0.25, applied to q AND k

#define MODE_Q 0
#define MODE_K 1
#define MODE_V 2
#define MODE_O 3

#if defined(__has_builtin)
#if __has_builtin(__builtin_amdgcn_global_load_async_to_lds_b128)
#define HAVE_ASYNC_LDS 1
#endif
#endif

#ifdef HAVE_ASYNC_LDS
#if __has_builtin(__builtin_amdgcn_s_wait_asynccnt)
#define WAIT_ASYNC0() __builtin_amdgcn_s_wait_asynccnt(0)
#else
#define WAIT_ASYNC0() asm volatile("s_wait_asynccnt 0x0" ::: "memory")
#endif
static __device__ __forceinline__ void async_copy16(void* lds, const void* gptr) {
  __builtin_amdgcn_global_load_async_to_lds_b128(
      (__attribute__((address_space(1))) v4i*)gptr,
      (__attribute__((address_space(3))) v4i*)lds, 0, 0);
}
#endif

static __device__ __forceinline__ v16h cat8(v8h lo, v8h hi) {
  return __builtin_shufflevector(lo, hi, 0,1,2,3,4,5,6,7,8,9,10,11,12,13,14,15);
}

// ---------------------------------------------------------------------------
// Generic f16 WMMA GEMM:  C[z][m][n] (+)= sum_k A[z][m+shift][k] * Bw[z][n][k]
// Block tile 64(M) x 64(N), 8 waves (4x2), each wave -> 16x32 of C (2 WMMA),
// K step 32. shift/seq implement conv1d taps (rows outside [0,seq) = 0).
// ---------------------------------------------------------------------------
__global__ __launch_bounds__(256)
void gemm_f16_wmma(const _Float16* __restrict__ A,
                   const _Float16* __restrict__ Bw,
                   float* __restrict__ C,
                   const float* __restrict__ bias,
                   int M, int N, int K,
                   int lda, int ldb, int ldc,
                   long long strideA, long long strideB, long long strideC,
                   int shift, int seq, int accumulate)
{
  __shared__ __attribute__((aligned(16))) _Float16 sA[2][64][40];
  __shared__ __attribute__((aligned(16))) _Float16 sB[2][64][40];

  const int t    = threadIdx.x;
  const int lane = t & 31;
  const int wave = t >> 5;
  const int wr   = wave >> 1;   // 0..3 : 16-row slab
  const int wc   = wave & 1;    // 0..1 : 32-col slab
  const int z    = blockIdx.z;

  A  += (long long)z * strideA;
  Bw += (long long)z * strideB;
  C  += (long long)z * strideC;

  const int m0 = blockIdx.x * 64;
  const int n0 = blockIdx.y * 64;

  // A staging: 64 rows x 32 cols, one 16B chunk per thread
  const int ar = t >> 2;
  const int ac = (t & 3) * 8;
  const int am = m0 + ar;
  const int as = (am % seq) + shift;
  const bool avalid = (unsigned)as < (unsigned)seq;
  const _Float16* aptr = A + (long long)(am + shift) * lda + ac;

  // B staging: 64 rows x 32 cols, one 16B chunk per thread
  const int bn = t >> 2;
  const int bc = (t & 3) * 8;
  const _Float16* bptr = Bw + (long long)(n0 + bn) * ldb + bc;

  const int r    = lane & 15;
  const int half = lane >> 4;

  v8f acc0 = {};
  v8f acc1 = {};

#ifdef HAVE_ASYNC_LDS
  // Conv-shift rows outside the sequence contribute zeros: fill once, async
  // copies below are EXEC-masked off for those lanes and never touch them.
  if (!avalid) {
    v8h zz = {};
    *(v8h*)&sA[0][ar][ac] = zz;
    *(v8h*)&sA[1][ar][ac] = zz;
  }
  // Prologue: tile 0 in flight
  if (avalid) async_copy16(&sA[0][ar][ac], aptr);
  async_copy16(&sB[0][bn][bc], bptr);

  int buf = 0;
  for (int k0 = 0; k0 < K; k0 += 32) {
    WAIT_ASYNC0();          // this wave's tile-k copies have landed in LDS
    __syncthreads();        // everyone's have

    if (k0 + 32 < K) {      // start tile k+1 while computing tile k
      if (avalid) async_copy16(&sA[buf ^ 1][ar][ac], aptr + k0 + 32);
      async_copy16(&sB[buf ^ 1][bn][bc], bptr + k0 + 32);
    }

    v8h alo = *(v8h*)&sA[buf][wr * 16 + r][half * 16];
    v8h ahi = *(v8h*)&sA[buf][wr * 16 + r][half * 16 + 8];
    v16h af = cat8(alo, ahi);

    v8h b0lo = *(v8h*)&sB[buf][wc * 32 + r][half * 16];
    v8h b0hi = *(v8h*)&sB[buf][wc * 32 + r][half * 16 + 8];
    v16h bf0 = cat8(b0lo, b0hi);
    acc0 = __builtin_amdgcn_wmma_f32_16x16x32_f16(
        false, af, false, bf0, (short)0, acc0, false, false);

    v8h b1lo = *(v8h*)&sB[buf][wc * 32 + 16 + r][half * 16];
    v8h b1hi = *(v8h*)&sB[buf][wc * 32 + 16 + r][half * 16 + 8];
    v16h bf1 = cat8(b1lo, b1hi);
    acc1 = __builtin_amdgcn_wmma_f32_16x16x32_f16(
        false, af, false, bf1, (short)0, acc1, false, false);

    buf ^= 1;
  }
#else
  for (int k0 = 0; k0 < K; k0 += 32) {
    v8h av = {};
    if (avalid) av = *(const v8h*)(aptr + k0);
    *(v8h*)&sA[0][ar][ac] = av;
    *(v8h*)&sB[0][bn][bc] = *(const v8h*)(bptr + k0);
    __syncthreads();

    if (avalid && (k0 + 32 < K)) __builtin_prefetch(aptr + k0 + 32, 0, 3);

    v8h alo = *(v8h*)&sA[0][wr * 16 + r][half * 16];
    v8h ahi = *(v8h*)&sA[0][wr * 16 + r][half * 16 + 8];
    v16h af = cat8(alo, ahi);

    v8h b0lo = *(v8h*)&sB[0][wc * 32 + r][half * 16];
    v8h b0hi = *(v8h*)&sB[0][wc * 32 + r][half * 16 + 8];
    v16h bf0 = cat8(b0lo, b0hi);
    acc0 = __builtin_amdgcn_wmma_f32_16x16x32_f16(
        false, af, false, bf0, (short)0, acc0, false, false);

    v8h b1lo = *(v8h*)&sB[0][wc * 32 + 16 + r][half * 16];
    v8h b1hi = *(v8h*)&sB[0][wc * 32 + 16 + r][half * 16 + 8];
    v16h bf1 = cat8(b1lo, b1hi);
    acc1 = __builtin_amdgcn_wmma_f32_16x16x32_f16(
        false, af, false, bf1, (short)0, acc1, false, false);
    __syncthreads();
  }
#endif

  // C tiles 16x16: VGPR j -> M = j + 8*(lane>>4); col = lane&15
  const int col0   = n0 + wc * 32 + r;
  const int col1   = col0 + 16;
  const int rowTop = m0 + wr * 16 + half * 8;
  const float bv0 = bias ? bias[col0] : 0.0f;
  const float bv1 = bias ? bias[col1] : 0.0f;
#pragma unroll
  for (int j = 0; j < 8; j++) {
    const long long row = rowTop + j;
    long long i0 = row * ldc + col0;
    long long i1 = row * ldc + col1;
    float v0 = acc0[j] + bv0;
    float v1 = acc1[j] + bv1;
    if (accumulate) { v0 += C[i0]; v1 += C[i1]; }
    C[i0] = v0;
    C[i1] = v1;
  }
}

// ---------------------------------------------------------------------------
// Fused online-softmax(qk) @ V, WMMA. One wave per 16 query rows; 4 f32
// accumulators cover dh=64 columns; w tile built in LDS per 32-col chunk.
// vT is [B,H,dh,S] f16 so it is directly the WMMA B operand (col=d, k=n).
// ---------------------------------------------------------------------------
__global__ __launch_bounds__(256)
void softmax_wv_wmma(const float* __restrict__ qk,
                     const _Float16* __restrict__ vT,
                     _Float16* __restrict__ wvh)
{
  __shared__ __attribute__((aligned(16))) _Float16 sW[8][16][40];

  const int t    = threadIdx.x;
  const int lane = t & 31;
  const int w    = t >> 5;
  const int z    = blockIdx.y;            // b*H + h
  const int rowBase = blockIdx.x * 128 + w * 16;
  const int r    = lane & 15;
  const int half = lane >> 4;

  const float* qrow = qk + ((long long)z * SEQ + rowBase + r) * SEQ;

  // Pass 1: online max/sum over this row (each lane owns half the row)
  float m = -3.0e38f, s = 0.0f;
  for (int j = 0; j < 512; j++) {
    float v  = qrow[half * 512 + j];
    float nm = fmaxf(m, v);
    s = s * __expf(m - nm) + __expf(v - nm);
    m = nm;
  }
  { // combine the two half-row lanes (xor 16)
    float mo = __shfl_xor(m, 16);
    float so = __shfl_xor(s, 16);
    float nm = fmaxf(m, mo);
    s = s * __expf(m - nm) + so * __expf(mo - nm);
    m = nm;
  }
  const float inv = 1.0f / s;

  v8f acc[4] = {v8f{}, v8f{}, v8f{}, v8f{}};

  // Pass 2: build w tiles (16x32 f16) in LDS, WMMA against vT chunks
  for (int kc = 0; kc < 32; kc++) {
    const int cb = half * 16;
#pragma unroll
    for (int i = 0; i < 16; i++) {
      float v = __expf(qrow[kc * 32 + cb + i] - m) * inv;
      sW[w][r][cb + i] = (_Float16)v;
    }
    __syncthreads();

    v8h alo = *(v8h*)&sW[w][r][half * 16];
    v8h ahi = *(v8h*)&sW[w][r][half * 16 + 8];
    v16h af = cat8(alo, ahi);
#pragma unroll
    for (int c = 0; c < 4; c++) {
      const int d = c * 16 + r;
      const _Float16* bp =
          vT + ((long long)z * DHEAD + d) * SEQ + kc * 32 + half * 16;
      v8h blo = *(const v8h*)bp;
      v8h bhi = *(const v8h*)(bp + 8);
      v16h bf = cat8(blo, bhi);
      acc[c] = __builtin_amdgcn_wmma_f32_16x16x32_f16(
          false, af, false, bf, (short)0, acc[c], false, false);
    }
    __syncthreads();
  }

  // Store wv as f16 [b, s, h*64+d]
  const int b  = z >> 4;
  const int hh = z & 15;
#pragma unroll
  for (int c = 0; c < 4; c++) {
    const int d = c * 16 + r;
#pragma unroll
    for (int j = 0; j < 8; j++) {
      const int srow = rowBase + half * 8 + j;
      wvh[((long long)(b * SEQ + srow)) * D_MODEL + hh * DHEAD + d] =
          (_Float16)acc[c][j];
    }
  }
}

// ---------------------------------------------------------------------------
// Small support kernels
// ---------------------------------------------------------------------------
__global__ void cast_f32_to_f16(const float* __restrict__ in,
                                _Float16* __restrict__ out, int n) {
  int i = blockIdx.x * blockDim.x + threadIdx.x;
  if (i < n) out[i] = (_Float16)in[i];
}

// out[t][o][i] = conv_w[o][i][t]   (conv_w is [D, D, 3])
__global__ void conv_repack_f16(const float* __restrict__ w,
                                _Float16* __restrict__ out, int n) {
  int i = blockIdx.x * blockDim.x + threadIdx.x;
  if (i >= n) return;
  int tap = i / (D_MODEL * D_MODEL);
  int rem = i - tap * (D_MODEL * D_MODEL);
  int o = rem / D_MODEL, ii = rem - o * D_MODEL;
  out[i] = (_Float16)w[(o * D_MODEL + ii) * 3 + tap];
}

__global__ void router_scores(const float* __restrict__ xc,
                              const float* __restrict__ pw,
                              const float* __restrict__ pb,
                              float* __restrict__ scores) {
  int mI = blockIdx.x * blockDim.x + threadIdx.x;
  if (mI >= BATCH * SEQ) return;
  const float* xr = xc + (long long)mI * D_MODEL;
  float acc = 0.0f;
  for (int d = 0; d < D_MODEL; d++) acc += xr[d] * pw[d];
  scores[mI] = acc + pb[0];
}

__global__ void router_softmax(const float* __restrict__ scores,
                               float* __restrict__ wsm) {
  __shared__ float red[256];
  int b = blockIdx.x, t = threadIdx.x;
  const float* sc = scores + b * SEQ;
  float lm = -3.0e38f;
  for (int i = t; i < SEQ; i += 256) lm = fmaxf(lm, sc[i]);
  red[t] = lm; __syncthreads();
  for (int o = 128; o > 0; o >>= 1) {
    if (t < o) red[t] = fmaxf(red[t], red[t + o]);
    __syncthreads();
  }
  float m = red[0]; __syncthreads();
  float ls = 0.0f;
  for (int i = t; i < SEQ; i += 256) ls += __expf(sc[i] - m);
  red[t] = ls; __syncthreads();
  for (int o = 128; o > 0; o >>= 1) {
    if (t < o) red[t] += red[t + o];
    __syncthreads();
  }
  float inv = 1.0f / red[0];
  for (int i = t; i < SEQ; i += 256) wsm[b * SEQ + i] = __expf(sc[i] - m) * inv;
}

__global__ void router_pool(const float* __restrict__ wsm,
                            const float* __restrict__ xc,
                            float* __restrict__ pooled) {
  int tid = blockIdx.x * blockDim.x + threadIdx.x;   // B*D
  if (tid >= BATCH * D_MODEL) return;
  int b = tid >> 10, d = tid & (D_MODEL - 1);
  float acc = 0.0f;
  for (int s = 0; s < SEQ; s++)
    acc += wsm[b * SEQ + s] * xc[((long long)(b * SEQ + s)) * D_MODEL + d];
  pooled[tid] = acc;
}

__global__ void router_route(const float* __restrict__ pooled,
                             const float* __restrict__ rw,
                             const float* __restrict__ rb,
                             float* __restrict__ route) {
  int t = threadIdx.x;               // 32 threads: (b,e)
  int b = t >> 3, e = t & 7;
  const float* pr = pooled + b * D_MODEL;
  const float* wr = rw + e * D_MODEL;
  float l = rb[e];
  for (int d = 0; d < D_MODEL; d++) l += pr[d] * wr[d];
  float m = l;
  for (int o = 1; o < 8; o <<= 1) m = fmaxf(m, __shfl_xor(m, o, 32));
  float ex = __expf(l - m);
  float s = ex;
  for (int o = 1; o < 8; o <<= 1) s += __shfl_xor(s, o, 32);
  route[t] = ex / s;
}

// h[b,e,s,r] = dot(inp[b,s,:], A[e,r,:])
__global__ void lora_h(const _Float16* __restrict__ inp,
                       const float* __restrict__ Am,
                       float* __restrict__ h) {
  int tid = blockIdx.x * blockDim.x + threadIdx.x;   // B*E*S*R = 262144
  if (tid >= BATCH * NEXP * SEQ * RANK) return;
  int rr = tid & 7;
  int s  = (tid >> 3) & (SEQ - 1);
  int e  = (tid >> 13) & 7;
  int b  = tid >> 16;
  const _Float16* xr = inp + ((long long)(b * SEQ + s)) * D_MODEL;
  const float*    ar = Am + ((long long)(e * RANK + rr)) * D_MODEL;
  float acc = 0.0f;
  for (int d = 0; d < D_MODEL; d++) acc += (float)xr[d] * ar[d];
  h[tid] = acc;
}

// proj + LORA_SCALE * sum_e route[b,e] * h[b,e,s,:] . Bm[e,n,:]
// then scatter to f16 head layouts (Q/K scaled & row-major, V transposed)
// or fp32 final output (O).
__global__ void add_lora_epilogue(const float* __restrict__ proj,
                                  const float* __restrict__ h,
                                  const float* __restrict__ Bm,
                                  const float* __restrict__ route,
                                  _Float16* __restrict__ out16,
                                  float* __restrict__ out32,
                                  int mode) {
  int tid = blockIdx.x * blockDim.x + threadIdx.x;   // B*S*D
  if (tid >= BATCH * SEQ * D_MODEL) return;
  int n = tid & (D_MODEL - 1);
  int s = (tid >> 10) & (SEQ - 1);
  int b = tid >> 20;
  float lora = 0.0f;
#pragma unroll
  for (int e = 0; e < NEXP; e++) {
    const float* hr = h + ((long long)((b * NEXP + e) * SEQ + s)) * RANK;
    const float* br = Bm + ((long long)(e * D_MODEL + n)) * RANK;
    float ta = 0.0f;
#pragma unroll
    for (int r2 = 0; r2 < RANK; r2++) ta += hr[r2] * br[r2];
    lora += route[b * NEXP + e] * ta;
  }
  float val = proj[tid] + LORA_SCALE * lora;
  int hh = n >> 6, dd = n & 63;
  if (mode == MODE_Q || mode == MODE_K) {
    val *= QK_SCALE;
    out16[(((long long)(b * NHEAD + hh)) * SEQ + s) * DHEAD + dd] = (_Float16)val;
  } else if (mode == MODE_V) {
    out16[(((long long)(b * NHEAD + hh)) * DHEAD + dd) * SEQ + s] = (_Float16)val;
  } else {
    out32[tid] = val;
  }
}

// ---------------------------------------------------------------------------
// Host-side orchestration
// ---------------------------------------------------------------------------
extern "C" void kernel_launch(void* const* d_in, const int* in_sizes, int n_in,
                              void* d_out, int out_size, void* d_ws, size_t ws_size,
                              hipStream_t stream) {
  (void)in_sizes; (void)n_in; (void)out_size; (void)ws_size;
  const float* x      = (const float*)d_in[0];
  const float* conv_w = (const float*)d_in[1];
  const float* conv_b = (const float*)d_in[2];
  const float* pool_w = (const float*)d_in[3];
  const float* pool_b = (const float*)d_in[4];
  const float* rlin_w = (const float*)d_in[5];
  const float* rlin_b = (const float*)d_in[6];
  const float* qW = (const float*)d_in[7];
  const float* qb = (const float*)d_in[8];
  const float* qA = (const float*)d_in[9];
  const float* qB = (const float*)d_in[10];
  const float* kW = (const float*)d_in[11];
  const float* kA = (const float*)d_in[12];
  const float* kB = (const float*)d_in[13];
  const float* vW = (const float*)d_in[14];
  const float* vb = (const float*)d_in[15];
  const float* vA = (const float*)d_in[16];
  const float* vB = (const float*)d_in[17];
  const float* oW = (const float*)d_in[18];
  const float* ob = (const float*)d_in[19];
  const float* oA = (const float*)d_in[20];
  const float* oB = (const float*)d_in[21];

  float* out = (float*)d_out;                                   // [B,S,D]
  float* qk  = out + (size_t)BATCH * SEQ * D_MODEL;             // [B,H,S,S]

  // workspace carve-out (256B aligned)
  char* wsp = (char*)d_ws;
  auto carve = [&](size_t bytes) {
    char* p = wsp;
    wsp += (bytes + 255) & ~(size_t)255;
    return p;
  };
  const size_t BSD = (size_t)BATCH * SEQ * D_MODEL;             // 4M elems
  const size_t DD  = (size_t)D_MODEL * D_MODEL;                 // 1M elems

  _Float16* xh    = (_Float16*)carve(BSD * 2);                  // x in f16
  _Float16* wtap  = (_Float16*)carve(3 * DD * 2);               // conv taps
  _Float16* wqh   = (_Float16*)carve(DD * 2);
  _Float16* wkh   = (_Float16*)carve(DD * 2);
  _Float16* wvh_w = (_Float16*)carve(DD * 2);
  _Float16* woh   = (_Float16*)carve(DD * 2);
  float*    xc    = (float*)carve(BSD * 4);                     // conv output
  float*    scores= (float*)carve((size_t)BATCH * SEQ * 4);
  float*    wsm   = (float*)carve((size_t)BATCH * SEQ * 4);
  float*    pooled= (float*)carve((size_t)BATCH * D_MODEL * 4);
  float*    route = (float*)carve((size_t)BATCH * NEXP * 4);
  float*    hbuf  = (float*)carve((size_t)BATCH * NEXP * SEQ * RANK * 4);
  float*    proj  = (float*)carve(BSD * 4);                     // reused q/k/v/o
  _Float16* qh    = (_Float16*)carve(BSD * 2);                  // [B,H,S,dh]
  _Float16* kh    = (_Float16*)carve(BSD * 2);                  // [B,H,S,dh]
  _Float16* vT    = (_Float16*)carve(BSD * 2);                  // [B,H,dh,S]
  _Float16* wvh   = (_Float16*)carve(BSD * 2);                  // [B,S,D]

  const int BLK = 256;

  // 1) casts + conv weight repack
  cast_f32_to_f16<<<(int)(BSD / BLK), BLK, 0, stream>>>(x, xh, (int)BSD);
  cast_f32_to_f16<<<(int)(DD / BLK), BLK, 0, stream>>>(qW, wqh, (int)DD);
  cast_f32_to_f16<<<(int)(DD / BLK), BLK, 0, stream>>>(kW, wkh, (int)DD);
  cast_f32_to_f16<<<(int)(DD / BLK), BLK, 0, stream>>>(vW, wvh_w, (int)DD);
  cast_f32_to_f16<<<(int)(DD / BLK), BLK, 0, stream>>>(oW, woh, (int)DD);
  conv_repack_f16<<<(int)(3 * DD / BLK), BLK, 0, stream>>>(conv_w, wtap, (int)(3 * DD));

  // 2) conv1d as 3 shifted WMMA GEMMs accumulated into xc
  dim3 gProj(4096 / 64, D_MODEL / 64, 1);
  for (int tap = 0; tap < 3; tap++) {
    gemm_f16_wmma<<<gProj, BLK, 0, stream>>>(
        xh, wtap + (size_t)tap * DD, xc, (tap == 0) ? conv_b : nullptr,
        4096, D_MODEL, D_MODEL, D_MODEL, D_MODEL, D_MODEL,
        0, 0, 0, tap - 1, SEQ, (tap == 0) ? 0 : 1);
  }

  // 3) router
  router_scores<<<(BATCH * SEQ) / BLK, BLK, 0, stream>>>(xc, pool_w, pool_b, scores);
  router_softmax<<<BATCH, BLK, 0, stream>>>(scores, wsm);
  router_pool<<<(BATCH * D_MODEL) / BLK, BLK, 0, stream>>>(wsm, xc, pooled);
  router_route<<<1, 32, 0, stream>>>(pooled, rlin_w, rlin_b, route);

  const int nH   = BATCH * NEXP * SEQ * RANK;   // 262144
  const int nBSD = (int)BSD;                    // 4194304

  // 4) Q / K / V projections (base WMMA GEMM + LoRA epilogue -> f16 heads)
  lora_h<<<nH / BLK, BLK, 0, stream>>>(xh, qA, hbuf);
  gemm_f16_wmma<<<gProj, BLK, 0, stream>>>(xh, wqh, proj, qb,
      4096, D_MODEL, D_MODEL, D_MODEL, D_MODEL, D_MODEL, 0, 0, 0, 0, SEQ, 0);
  add_lora_epilogue<<<nBSD / BLK, BLK, 0, stream>>>(proj, hbuf, qB, route, qh, nullptr, MODE_Q);

  lora_h<<<nH / BLK, BLK, 0, stream>>>(xh, kA, hbuf);
  gemm_f16_wmma<<<gProj, BLK, 0, stream>>>(xh, wkh, proj, nullptr,
      4096, D_MODEL, D_MODEL, D_MODEL, D_MODEL, D_MODEL, 0, 0, 0, 0, SEQ, 0);
  add_lora_epilogue<<<nBSD / BLK, BLK, 0, stream>>>(proj, hbuf, kB, route, kh, nullptr, MODE_K);

  lora_h<<<nH / BLK, BLK, 0, stream>>>(xh, vA, hbuf);
  gemm_f16_wmma<<<gProj, BLK, 0, stream>>>(xh, wvh_w, proj, vb,
      4096, D_MODEL, D_MODEL, D_MODEL, D_MODEL, D_MODEL, 0, 0, 0, 0, SEQ, 0);
  add_lora_epilogue<<<nBSD / BLK, BLK, 0, stream>>>(proj, hbuf, vB, route, vT, nullptr, MODE_V);

  // 5) qk logits [B,H,S,S] -> d_out (z-batched over 64 (b,h) slices)
  dim3 gQK(SEQ / 64, SEQ / 64, BATCH * NHEAD);
  gemm_f16_wmma<<<gQK, BLK, 0, stream>>>(
      qh, kh, qk, nullptr, SEQ, SEQ, DHEAD, DHEAD, DHEAD, SEQ,
      (long long)SEQ * DHEAD, (long long)SEQ * DHEAD, (long long)SEQ * SEQ,
      0, SEQ, 0);

  // 6) fused online softmax + w@V
  dim3 gSW(SEQ / 128, BATCH * NHEAD, 1);
  softmax_wv_wmma<<<gSW, BLK, 0, stream>>>(qk, vT, wvh);

  // 7) output projection (+ LoRA) -> d_out
  lora_h<<<nH / BLK, BLK, 0, stream>>>(wvh, oA, hbuf);
  gemm_f16_wmma<<<gProj, BLK, 0, stream>>>(wvh, woh, proj, ob,
      4096, D_MODEL, D_MODEL, D_MODEL, D_MODEL, D_MODEL, 0, 0, 0, 0, SEQ, 0);
  add_lora_epilogue<<<nBSD / BLK, BLK, 0, stream>>>(proj, hbuf, oB, route, nullptr, out, MODE_O);
}
